// InteractionBlock_11192684773936
// MI455X (gfx1250) — matrix operand order repaired and checked by reading, not compiled, not run on hardware
//
#include <hip/hip_runtime.h>
#include <hip/hip_bf16.h>

// ---------------------------------------------------------------------------
// CDNA5 (gfx1250) InteractionBlock.
//  - All GEMMs: v_wmma_f32_16x16x32_bf16 (wave32), f32 accumulate.
//  - Weights pre-packed into per-lane B-fragment layout (bf16). Per k-step,
//    all 8 B fragments are clause-loaded up front, then the wmma chain drains
//    them with incremental loadcnt waits (in-wave latency overlap).
//  - f32 -> bf16 uses native v_cvt_pk_bf16_f32.
//  - Intermediate activations stored bf16 (consumed as WMMA A operands; x1 also scales edge messages).
//  - Edge message * gather + segment-sum fused into edge kernel epilogue via
//    global_atomic_add_f32 (node working set is L2-resident).
// ---------------------------------------------------------------------------

typedef __attribute__((ext_vector_type(16))) __bf16          v16bf;
typedef __attribute__((ext_vector_type(2)))  __bf16          v2bf;
typedef __attribute__((ext_vector_type(8)))  float           v8f;
typedef __attribute__((ext_vector_type(8)))  unsigned short  v8us;

#define NND   20000
#define NEDGE 640000
#define HDIM  128

union BF16Frag {
    v16bf v;
    unsigned short u[16];
    unsigned int   w[8];
    v8us h[2];
};

// native scalar convert (RNE)
__device__ __forceinline__ unsigned short f32_to_bf16_rne(float f) {
    __bf16 h = (__bf16)f;
    return __builtin_bit_cast(unsigned short, h);
}

// native packed convert of a float pair -> one dword of 2 x bf16
__device__ __forceinline__ unsigned int cvt_pk_bf16(float a, float b) {
    v2bf t;
    t.x = (__bf16)a;
    t.y = (__bf16)b;
    return __builtin_bit_cast(unsigned int, t);
}

__device__ __forceinline__ float bf16_to_f32(unsigned short u) {
    return __uint_as_float(((unsigned int)u) << 16);
}

__device__ __forceinline__ float swish_f(float v) {
    return v / (1.0f + __expf(-v));
}

// ---------------------------------------------------------------------------
// Zero-fill (graph-capture safe), vectorized.
// ---------------------------------------------------------------------------
__global__ void zero_kernel(float4* __restrict__ p, long n4) {
    long i = (long)blockIdx.x * blockDim.x + threadIdx.x;
    long stride = (long)gridDim.x * blockDim.x;
    const float4 z = {0.f, 0.f, 0.f, 0.f};
    for (; i < n4; i += stride) p[i] = z;
}

// ---------------------------------------------------------------------------
// Pack a row-major f32 weight [K, ldw] into per-lane WMMA B-fragment layout:
//   out[((ks*ntiles + t)*32 + lane)*16 + i] =
//       bf16( W[(ks*32 + (i>>3)*16 + (lane>>4)*8 + (i&7))*ldw + t*16 + (lane&15)] )   (0 pad)
// ---------------------------------------------------------------------------
__global__ void pack_weight_kernel(const float* __restrict__ W, int K, int ldw,
                                   int ntiles, unsigned short* __restrict__ out,
                                   int elems) {
    int idx = blockIdx.x * blockDim.x + threadIdx.x;
    if (idx >= elems) return;
    const int i    = idx & 15;
    const int lane = (idx >> 4) & 31;
    const int t    = (idx >> 9) % ntiles;
    const int ks   = (idx >> 9) / ntiles;
    const int k    = ks * 32 + ((i >> 3) << 4) + ((lane >> 4) << 3) + (i & 7);
    const int col  = t * 16 + (lane & 15);
    out[idx] = (k < K) ? f32_to_bf16_rne(W[(size_t)k * ldw + col])
                       : (unsigned short)0;
}

// ---------------------------------------------------------------------------
// Generic WMMA bf16 GEMM (compile-time specialized):
//   out[M,128] = act( A[M,K]@W (+ A2[M,K]@W2) + bias ),  K = KSTEPS*32
// ABF : A dtype (0 = f32, 1 = bf16);  A2M : 0 = no second product, 1 = bf16 A2
// ACT : 1 = swish;  OUTB : output dtype (0 = f32, 1 = bf16)
// One wave per 16-row stripe, 8 column tiles. M must be a multiple of 16.
// ---------------------------------------------------------------------------
template <int ABF, int A2M, int ACT, int OUTB, int KSTEPS>
__global__ void wmma_gemm_kernel(const void* __restrict__ A, int lda,
                                 const unsigned short* __restrict__ Wp,
                                 const void* __restrict__ A2,
                                 const unsigned short* __restrict__ W2p,
                                 const float* __restrict__ bias,
                                 void* __restrict__ out, int ldout, int M)
{
    const int wave = threadIdx.x >> 5;
    const int lane = threadIdx.x & 31;
    const int rowBase = blockIdx.x * 64 + wave * 16;
    if (rowBase >= M) return;

    const int r  = lane & 15;
    const int hi = lane >> 4;
    const int hb = hi * 8;

    v8f acc[8];
#pragma unroll
    for (int t = 0; t < 8; ++t) acc[t] = {};

    const size_t rowOff = (size_t)(rowBase + r) * lda;
    const float*          ArowF  = (const float*)A + rowOff;           // ABF==0
    const unsigned short* ArowH  = (const unsigned short*)A + rowOff;  // ABF==1
    const unsigned short* A2rowH = (const unsigned short*)A2 + rowOff; // A2M==1

    const v16bf* Wv  = (const v16bf*)Wp;
    const v16bf* W2v = (const v16bf*)W2p;

#pragma unroll
    for (int ks = 0; ks < KSTEPS; ++ks) {
        const int k0 = ks * 32;
        // ---- A fragment(s) ----
        BF16Frag a, a2;
        if constexpr (ABF) {
            a.h[0] = *(const v8us*)(ArowH + k0 + hb);
            a.h[1] = *(const v8us*)(ArowH + k0 + 16 + hb);
        } else {
            const float4 f0 = *(const float4*)(ArowF + k0 + hb);
            const float4 f1 = *(const float4*)(ArowF + k0 + hb + 4);
            const float4 f2 = *(const float4*)(ArowF + k0 + 16 + hb);
            const float4 f3 = *(const float4*)(ArowF + k0 + 16 + hb + 4);
            a.w[0] = cvt_pk_bf16(f0.x, f0.y); a.w[1] = cvt_pk_bf16(f0.z, f0.w);
            a.w[2] = cvt_pk_bf16(f1.x, f1.y); a.w[3] = cvt_pk_bf16(f1.z, f1.w);
            a.w[4] = cvt_pk_bf16(f2.x, f2.y); a.w[5] = cvt_pk_bf16(f2.z, f2.w);
            a.w[6] = cvt_pk_bf16(f3.x, f3.y); a.w[7] = cvt_pk_bf16(f3.z, f3.w);
        }
        if constexpr (A2M) {
            a2.h[0] = *(const v8us*)(A2rowH + k0 + hb);
            a2.h[1] = *(const v8us*)(A2rowH + k0 + 16 + hb);
        }

        if constexpr (!A2M) {
            // preload all 8 B fragments (one big clause), then drain with wmma
            v16bf b[8];
#pragma unroll
            for (int t = 0; t < 8; ++t)
                b[t] = Wv[(size_t)(ks * 8 + t) * 32 + lane];
#pragma unroll
            for (int t = 0; t < 8; ++t)
                acc[t] = __builtin_amdgcn_wmma_f32_16x16x32_bf16(
                    false, a.v, false, b[t], (short)0, acc[t], false, false);
        } else {
            // dual product: interleave to cap register pressure
#pragma unroll
            for (int t = 0; t < 8; ++t) {
                v16bf b = Wv[(size_t)(ks * 8 + t) * 32 + lane];
                acc[t] = __builtin_amdgcn_wmma_f32_16x16x32_bf16(
                    false, a.v, false, b, (short)0, acc[t], false, false);
                v16bf b2 = W2v[(size_t)(ks * 8 + t) * 32 + lane];
                acc[t] = __builtin_amdgcn_wmma_f32_16x16x32_bf16(
                    false, a2.v, false, b2, (short)0, acc[t], false, false);
            }
        }
    }

    // ---- epilogue: bias, swish, store (C: M = v + 8*hi, N = lane&15) ----
    float*          outF = (float*)out;
    unsigned short* outH = (unsigned short*)out;
#pragma unroll
    for (int t = 0; t < 8; ++t) {
        const int col = t * 16 + r;
        const float bv = bias[col];
#pragma unroll
        for (int v = 0; v < 8; ++v) {
            const int mrow = rowBase + v + 8 * hi;     // M % 16 == 0: in range
            float val = acc[t][v] + bv;
            if constexpr (ACT) val = swish_f(val);
            const size_t o = (size_t)mrow * ldout + col;
            if constexpr (OUTB) outH[o] = f32_to_bf16_rne(val);
            else                outF[o] = val;
        }
    }
}

// ---------------------------------------------------------------------------
// Edge-feature A-fragment loader: 8 elements at K = KB + hb + i from a row of
// F floats (zero-padded past F). Full groups -> two aligned float4 loads +
// packed converts.
// ---------------------------------------------------------------------------
template <int F, int KB>
__device__ __forceinline__ void load_a8(BF16Frag& a, int idx0,
                                        const float* __restrict__ frow, int hb) {
    if constexpr (KB + 15 < F) {               // both half-wave variants in range
        const float4 f0 = *(const float4*)(frow + KB + hb);
        const float4 f1 = *(const float4*)(frow + KB + hb + 4);
        const int j = idx0 >> 1;
        a.w[j + 0] = cvt_pk_bf16(f0.x, f0.y);
        a.w[j + 1] = cvt_pk_bf16(f0.z, f0.w);
        a.w[j + 2] = cvt_pk_bf16(f1.x, f1.y);
        a.w[j + 3] = cvt_pk_bf16(f1.z, f1.w);
    } else if constexpr (KB >= F) {            // fully padded
#pragma unroll
        for (int i = 0; i < 8; ++i) a.u[idx0 + i] = 0;
    } else {                                   // mixed (per-element select)
#pragma unroll
        for (int i = 0; i < 8; ++i) {
            const int k = KB + hb + i;
            a.u[idx0 + i] = (k < F) ? f32_to_bf16_rne(frow[k]) : (unsigned short)0;
        }
    }
}

// ---------------------------------------------------------------------------
// Fused edge kernel:  f = (feat[E,F] @ W0[F,64]) @ W1[64,128]   (bf16 WMMA)
// then per-edge atomicAdd(agg[dst], f * x1[src]).  E multiple of 64.
// ---------------------------------------------------------------------------
template <int F>
__global__ void edge_conv_kernel(const float* __restrict__ feat,
                                 const unsigned short* __restrict__ W0p, // [KS][4][32][16]
                                 const unsigned short* __restrict__ W1p, // [2][8][32][16]
                                 const unsigned short* __restrict__ x1h, // [N,128] bf16
                                 const int* __restrict__ src,
                                 const int* __restrict__ dst,
                                 float* __restrict__ agg)                // [N,128] f32
{
    __shared__ alignas(16) unsigned short tS[4][16][72];  // bf16 of t = feat@W0

    const int wave = threadIdx.x >> 5;
    const int lane = threadIdx.x & 31;
    const int eBase = blockIdx.x * 64 + wave * 16;
    const int r  = lane & 15;
    const int hi = lane >> 4;
    const int hb = hi * 8;

    const v16bf* W0v = (const v16bf*)W0p;
    const v16bf* W1v = (const v16bf*)W1p;

    // --- stage 1: t[16,64] = feat_tile @ W0 ---
    v8f acc1[4];
#pragma unroll
    for (int t = 0; t < 4; ++t) acc1[t] = {};

    const float* frow = feat + (size_t)(eBase + r) * F;
    __builtin_prefetch(feat + (size_t)(eBase + 64 + r) * F, 0, 1);

    {   // k-step 0
        BF16Frag a;
        load_a8<F, 0>(a, 0, frow, hb);
        load_a8<F, 16>(a, 8, frow, hb);
        v16bf b[4];
#pragma unroll
        for (int t = 0; t < 4; ++t) b[t] = W0v[(size_t)t * 32 + lane];
#pragma unroll
        for (int t = 0; t < 4; ++t)
            acc1[t] = __builtin_amdgcn_wmma_f32_16x16x32_bf16(
                false, a.v, false, b[t], (short)0, acc1[t], false, false);
    }
    if constexpr (F > 32) {   // k-step 1
        BF16Frag a;
        load_a8<F, 32>(a, 0, frow, hb);
        load_a8<F, 48>(a, 8, frow, hb);
        v16bf b[4];
#pragma unroll
        for (int t = 0; t < 4; ++t) b[t] = W0v[(size_t)(4 + t) * 32 + lane];
#pragma unroll
        for (int t = 0; t < 4; ++t)
            acc1[t] = __builtin_amdgcn_wmma_f32_16x16x32_bf16(
                false, a.v, false, b[t], (short)0, acc1[t], false, false);
    }

    // --- re-layout C -> A via LDS (bf16, 144B row stride for aligned b128) ---
#pragma unroll
    for (int t = 0; t < 4; ++t)
#pragma unroll
        for (int v = 0; v < 8; ++v)
            tS[wave][v + 8 * hi][t * 16 + r] = f32_to_bf16_rne(acc1[t][v]);
    __syncthreads();

    // --- stage 2: f[16,128] = t @ W1, K = 64 ---
    v8f acc2[8];
#pragma unroll
    for (int t = 0; t < 8; ++t) acc2[t] = {};

#pragma unroll
    for (int ks = 0; ks < 2; ++ks) {
        const int k0 = ks * 32;
        BF16Frag a;
        a.h[0] = *(const v8us*)&tS[wave][r][k0 + hb];        // ds_load_b128
        a.h[1] = *(const v8us*)&tS[wave][r][k0 + 16 + hb];
        v16bf b[8];
#pragma unroll
        for (int t = 0; t < 8; ++t) b[t] = W1v[(size_t)(ks * 8 + t) * 32 + lane];
#pragma unroll
        for (int t = 0; t < 8; ++t)
            acc2[t] = __builtin_amdgcn_wmma_f32_16x16x32_bf16(
                false, a.v, false, b[t], (short)0, acc2[t], false, false);
    }

    // --- fused message + scatter: agg[dst[e]] += f[e] * x1[src[e]] ---
#pragma unroll
    for (int v = 0; v < 8; ++v) {
        const int e = eBase + v + 8 * hi;
        const int s = src[e];
        const int d = dst[e];
        const unsigned short* xs = x1h + (size_t)s * HDIM;
        float* ad = agg + (size_t)d * HDIM;
#pragma unroll
        for (int t = 0; t < 8; ++t) {
            const int col = t * 16 + r;
            atomicAdd(&ad[col], acc2[t][v] * bf16_to_f32(xs[col]));
        }
    }
}

// ---------------------------------------------------------------------------
// Launcher
// ---------------------------------------------------------------------------
extern "C" void kernel_launch(void* const* d_in, const int* in_sizes, int n_in,
                              void* d_out, int out_size, void* d_ws, size_t ws_size,
                              hipStream_t stream) {
    (void)in_sizes; (void)n_in; (void)out_size; (void)ws_size;

    // ---- inputs (setup_inputs order; params = JAX pytree, dict keys sorted) --
    const float* x        = (const float*)d_in[0];   // [N,128]
    const float* feature0 = (const float*)d_in[1];   // [E,24]
    const float* feature1 = (const float*)d_in[2];   // [E,36]
    const float* pos_emb  = (const float*)d_in[3];   // [E,16]
    const int*   eidx     = (const int*)  d_in[4];   // [2,E]
    const float* cat0_w  = (const float*)d_in[6];   const float* cat0_b  = (const float*)d_in[7];
    const float* cat1_w  = (const float*)d_in[8];   const float* cat1_b  = (const float*)d_in[9];
    const float* cat2_w  = (const float*)d_in[10];  const float* cat2_b  = (const float*)d_in[11];
    const float* conv_wl[3] = {(const float*)d_in[12], (const float*)d_in[15], (const float*)d_in[18]};
    const float* conv_bl[3] = {(const float*)d_in[13], (const float*)d_in[16], (const float*)d_in[19]};
    const float* conv_wr[3] = {(const float*)d_in[14], (const float*)d_in[17], (const float*)d_in[20]};
    const float* fW0[3] = {(const float*)d_in[21], (const float*)d_in[23], (const float*)d_in[25]};
    const float* fW1[3] = {(const float*)d_in[22], (const float*)d_in[24], (const float*)d_in[26]};
    const float* final_w = (const float*)d_in[27];  const float* final_b = (const float*)d_in[28];
    const float* lin_w[3] = {(const float*)d_in[29], (const float*)d_in[31], (const float*)d_in[33]};
    const float* lin_b[3] = {(const float*)d_in[30], (const float*)d_in[32], (const float*)d_in[34]};
    const float* lin1_w = (const float*)d_in[35];   const float* lin1_b = (const float*)d_in[36];
    const float* lin2_w = (const float*)d_in[37];   const float* lin2_b = (const float*)d_in[38];
    const float* lins0_w = (const float*)d_in[39];  const float* lins0_b = (const float*)d_in[40];
    const float* lins1_w = (const float*)d_in[41];  const float* lins1_b = (const float*)d_in[42];

    const int* src = eidx;
    const int* dst = eidx + NEDGE;

    // ---- workspace layout ----
    const size_t NH = (size_t)NND * HDIM;                 // 2,560,000
    float* agg = (float*)d_ws;                            // 3*NH f32 (atomics)
    unsigned short* us = (unsigned short*)(agg + 3 * NH);
    unsigned short* x1h  = us + 0 * NH;                   // bf16 activations
    unsigned short* x2h  = us + 1 * NH;
    unsigned short* tmph = us + 2 * NH;
    unsigned short* bufA = us + 3 * NH;
    unsigned short* bufB = us + 4 * NH;
    unsigned short* hcat = us + 5 * NH;                   // [N,384]
    unsigned short* wp   = us + 8 * NH;                   // packed weights

    auto pack = [&](const float* W, int K, int ldw, unsigned short* dstp) {
        const int ntiles = ldw / 16;
        const int ksteps = (K + 31) / 32;
        const int elems  = ksteps * ntiles * 512;
        pack_weight_kernel<<<(elems + 255) / 256, 256, 0, stream>>>(
            W, K, ldw, ntiles, dstp, elems);
        return dstp + elems;
    };

    unsigned short* p = wp;
    unsigned short* lin1p = p;  p = pack(lin1_w, 128, 128, p);
    unsigned short* lin2p = p;  p = pack(lin2_w, 128, 128, p);
    unsigned short* wlP[3];  unsigned short* wrP[3];  unsigned short* linP[3];
    for (int c = 0; c < 3; ++c) { wlP[c]  = p; p = pack(conv_wl[c], 128, 128, p); }
    for (int c = 0; c < 3; ++c) { wrP[c]  = p; p = pack(conv_wr[c], 128, 128, p); }
    for (int c = 0; c < 3; ++c) { linP[c] = p; p = pack(lin_w[c],   128, 128, p); }
    unsigned short* cat0p = p;  p = pack(cat0_w, 384, 128, p);
    unsigned short* cat1p = p;  p = pack(cat1_w, 128, 128, p);
    unsigned short* cat2p = p;  p = pack(cat2_w, 128, 128, p);
    unsigned short* lins0p = p; p = pack(lins0_w, 128, 128, p);
    unsigned short* lins1p = p; p = pack(lins1_w, 128, 128, p);
    unsigned short* finalp = p; p = pack(final_w, 128, 128, p);
    const int Fdim[3] = {24, 36, 16};
    unsigned short* w0P[3]; unsigned short* w1P[3];
    for (int c = 0; c < 3; ++c) { w0P[c] = p; p = pack(fW0[c], Fdim[c], 64, p); }
    for (int c = 0; c < 3; ++c) { w1P[c] = p; p = pack(fW1[c], 64, 128, p); }

    const dim3 gBlk(128);
    const dim3 gGrid((NND + 63) / 64);            // 313
    const dim3 eGrid(NEDGE / 64);                 // 10000

    zero_kernel<<<2048, 256, 0, stream>>>((float4*)agg, (long)(3 * NH / 4));

    // x1 = swish(x@W1+b), x2 = swish(x@W2+b)   (A f32 -> out bf16)
    wmma_gemm_kernel<0,0,1,1,4><<<gGrid, gBlk, 0, stream>>>(
        x, HDIM, lin1p, nullptr, nullptr, lin1_b, x1h, HDIM, NND);
    wmma_gemm_kernel<0,0,1,1,4><<<gGrid, gBlk, 0, stream>>>(
        x, HDIM, lin2p, nullptr, nullptr, lin2_b, x2h, HDIM, NND);

    // edge features + fused message/scatter into agg_c
    edge_conv_kernel<24><<<eGrid, gBlk, 0, stream>>>(feature0, w0P[0], w1P[0],
                                                     x1h, src, dst, agg + 0 * NH);
    edge_conv_kernel<36><<<eGrid, gBlk, 0, stream>>>(feature1, w0P[1], w1P[1],
                                                     x1h, src, dst, agg + 1 * NH);
    edge_conv_kernel<16><<<eGrid, gBlk, 0, stream>>>(pos_emb,  w0P[2], w1P[2],
                                                     x1h, src, dst, agg + 2 * NH);

    // per-conv: tmp = agg@Wl + x1@Wr + bl ; h[:,128c] = swish(tmp@Wlin_c + b)
    for (int c = 0; c < 3; ++c) {
        wmma_gemm_kernel<0,1,0,1,4><<<gGrid, gBlk, 0, stream>>>(
            agg + (size_t)c * NH, HDIM, wlP[c], x1h, wrP[c], conv_bl[c],
            tmph, HDIM, NND);
        wmma_gemm_kernel<1,0,1,1,4><<<gGrid, gBlk, 0, stream>>>(
            tmph, HDIM, linP[c], nullptr, nullptr, lin_b[c],
            hcat + c * HDIM, 3 * HDIM, NND);
    }

    // cat MLP chain
    wmma_gemm_kernel<1,0,1,1,12><<<gGrid, gBlk, 0, stream>>>(
        hcat, 3 * HDIM, cat0p, nullptr, nullptr, cat0_b, bufA, HDIM, NND);
    wmma_gemm_kernel<1,0,1,1,4><<<gGrid, gBlk, 0, stream>>>(
        bufA, HDIM, cat1p, nullptr, nullptr, cat1_b, bufB, HDIM, NND);
    wmma_gemm_kernel<1,0,1,1,4><<<gGrid, gBlk, 0, stream>>>(
        bufB, HDIM, cat2p, nullptr, nullptr, cat2_b, bufA, HDIM, NND);

    // (h + x2) fused: (h+x2)@W = h@W + x2@W  (same packed W)
    wmma_gemm_kernel<1,1,1,1,4><<<gGrid, gBlk, 0, stream>>>(
        bufA, HDIM, lins0p, x2h, lins0p, lins0_b, bufB, HDIM, NND);
    wmma_gemm_kernel<1,0,1,1,4><<<gGrid, gBlk, 0, stream>>>(
        bufB, HDIM, lins1p, nullptr, nullptr, lins1_b, bufA, HDIM, NND);

    // final (no act) -> d_out f32
    wmma_gemm_kernel<1,0,0,0,4><<<gGrid, gBlk, 0, stream>>>(
        bufA, HDIM, finalp, nullptr, nullptr, final_b, d_out, HDIM, NND);
}
